// RNN_3341484556375
// MI455X (gfx1250) — matrix operand order repaired and checked by reading, not compiled, not run on hardware
//
#include <hip/hip_runtime.h>
#include <hip/hip_bf16.h>

typedef __attribute__((ext_vector_type(16))) _Float16 v16h;
typedef __attribute__((ext_vector_type(8)))  float    v8f;
typedef __attribute__((ext_vector_type(2)))  __fp16   fp16x2;

#define B_TOT 256
#define T_LEN 4096
#define H_DIM 32

__device__ __forceinline__ unsigned pack_h2(float a, float b) {
    union { fp16x2 p; unsigned u; } cv;
    cv.p = __builtin_amdgcn_cvt_pkrtz(a, b);   // v_cvt_pk_f16_f32 (RTZ)
    return cv.u;
}

// Branchless tanh: tanh(x) = 1 - 2/(exp(2x)+1).
// Exact identity; saturates correctly at +/-inf via exp overflow/underflow.
// Lowers to v_exp_f32 + v_rcp_f32 + 3 VALU, no EXEC branches.
__device__ __forceinline__ float fast_tanh(float x) {
    float e = __expf(x + x);
    float r = __builtin_amdgcn_rcpf(e + 1.0f);
    return __builtin_fmaf(-2.0f, r, 1.0f);
}

// One wave32 handles a 16-row batch tile for all T steps.
// Orientation: D[m,b] = sum_k W_hh[m,k] * h[b,k]  ( == (h @ W_hh^T)[b,m] )
// Row-permuted A operands so the D->next-B relayout is 100% lane-local:
//   A0 = W_hh rows {0..7, 16..23}   (M-pos 0..7, 8..15)
//   A1 = W_hh rows {8..15, 24..31}
// After WMMA, lane L (half = L/16) holds in acc0 components k+16*half and in
// acc1 components k+8+16*half  -> exactly the B-operand K-range e+16*half.
// No shuffles, no cndmasks in the recurrence chain.
__global__ __launch_bounds__(32) void rnn_wmma_kernel(
    const float* __restrict__ x,        // [B,T,1]
    const float* __restrict__ prev_h,   // [1,B,H]
    const float* __restrict__ W_ih,     // [H,1]
    const float* __restrict__ W_hh,     // [H,H] row-major
    const float* __restrict__ b_ih,     // [H]
    const float* __restrict__ b_hh,     // [H]
    const float* __restrict__ Wd,       // [1,H]
    const float* __restrict__ bd,       // [1]
    float* __restrict__ out)            // y [B,T] flat, then h_last [B,H]
{
    const int lane = threadIdx.x & 31;
    const int half = lane >> 4;   // which 16-lane half
    const int row  = lane & 15;   // batch row within tile (D's N column)
    const int b    = blockIdx.x * 16 + row;

    union V16 { v16h v; _Float16 h[16]; unsigned u[8]; };

    // ---- A operands: permuted W_hh rows in 16-bit A layout ----
    // A element i of lane: K = 16*half + i
    const int r0 = (row < 8) ? row       : row + 8;    // A0: rows 0..7,16..23
    const int r1 = (row < 8) ? row + 8   : row + 16;   // A1: rows 8..15,24..31
    V16 A0, A1;
    #pragma unroll
    for (int i = 0; i < 16; ++i) {
        int K = 16 * half + i;
        A0.h[i] = (_Float16)W_hh[r0 * H_DIM + K];
        A1.h[i] = (_Float16)W_hh[r1 * H_DIM + K];
    }

    // ---- per-lane coefficients in permuted D layout ----
    // acc0 vgpr k <-> component m0 = k + 16*half
    // acc1 vgpr k <-> component m1 = k + 8 + 16*half
    float wi0[8], wi1[8], bb0[8], bb1[8], wd0[8], wd1[8];
    #pragma unroll
    for (int k = 0; k < 8; ++k) {
        int m0 = k + 16 * half, m1 = k + 8 + 16 * half;
        wi0[k] = W_ih[m0];                 wi1[k] = W_ih[m1];
        bb0[k] = b_ih[m0] + b_hh[m0];      bb1[k] = b_ih[m1] + b_hh[m1];
        wd0[k] = Wd[m0];                   wd1[k] = Wd[m1];
    }
    const float bdv = bd[0];

    // ---- initial hidden state -> B operand layout ----
    // B lane L holds batch col = row, K = 16*half + e  (e = element index)
    V16 hB;
    {
        const float* hp = prev_h + (size_t)b * H_DIM + 16 * half;
        #pragma unroll
        for (int j = 0; j < 16; ++j) hB.h[j] = (_Float16)hp[j];
    }

    const float* xrow = x   + (size_t)b * T_LEN;
    float*       yrow = out + (size_t)b * T_LEN;

    // prefetch first 8 x values (contiguous in t for fixed b)
    float4 xn0 = ((const float4*)xrow)[0];
    float4 xn1 = ((const float4*)xrow)[1];

    float t0[8], t1[8];   // last step's tanh'd hidden (permuted D layout)

    for (int tc = 0; tc < T_LEN; tc += 8) {
        float xs[8];
        xs[0] = xn0.x; xs[1] = xn0.y; xs[2] = xn0.z; xs[3] = xn0.w;
        xs[4] = xn1.x; xs[5] = xn1.y; xs[6] = xn1.z; xs[7] = xn1.w;
        if (tc + 8 < T_LEN) {           // prefetch next chunk while we compute
            xn0 = ((const float4*)xrow)[(tc >> 2) + 2];
            xn1 = ((const float4*)xrow)[(tc >> 2) + 3];
        }

        float ybuf[8];
        #pragma unroll
        for (int j = 0; j < 8; ++j) {
            const float xv = xs[j];

            // C = xi in permuted D layout: xi[m] = x[b,t]*W_ih[m] + (b_ih+b_hh)[m]
            v8f c0, c1;
            #pragma unroll
            for (int k = 0; k < 8; ++k) {
                c0[k] = __builtin_fmaf(xv, wi0[k], bb0[k]);
                c1[k] = __builtin_fmaf(xv, wi1[k], bb1[k]);
            }

            // acc = W_hh_perm x h^T + xi  (two 16x16x32 f16 WMMAs)
            v8f acc0 = __builtin_amdgcn_wmma_f32_16x16x32_f16(
                false, A0.v, false, hB.v, (short)0, c0, false, false);
            v8f acc1 = __builtin_amdgcn_wmma_f32_16x16x32_f16(
                false, A1.v, false, hB.v, (short)0, c1, false, false);

            // elementwise branchless tanh in f32
            #pragma unroll
            for (int k = 0; k < 8; ++k) {
                t0[k] = fast_tanh(acc0[k]);
                t1[k] = fast_tanh(acc1[k]);
            }

            // rebuild B operand for next step: purely lane-local f16 packs
            // elements 0..7 = t0 (K = 16*half + 0..7), 8..15 = t1 (+8..15)
            #pragma unroll
            for (int q = 0; q < 4; ++q) {
                hB.u[q]     = pack_h2(t0[2 * q], t0[2 * q + 1]);
                hB.u[4 + q] = pack_h2(t1[2 * q], t1[2 * q + 1]);
            }

            // head: y = tanh(h . Wd + bd); lane holds comps 16*half..16*half+15,
            // partner lane holds the complement (off the recurrence chain)
            float s = 0.0f;
            #pragma unroll
            for (int k = 0; k < 8; ++k)
                s = __builtin_fmaf(t0[k], wd0[k], __builtin_fmaf(t1[k], wd1[k], s));
            s += __shfl_xor(s, 16, 32);
            ybuf[j] = fast_tanh(s + bdv);
        }

        if (half == 0) {   // one half-wave writes the y chunk (float4 x2)
            float4 y0 = { ybuf[0], ybuf[1], ybuf[2], ybuf[3] };
            float4 y1 = { ybuf[4], ybuf[5], ybuf[6], ybuf[7] };
            ((float4*)(yrow + tc))[0] = y0;
            ((float4*)(yrow + tc))[1] = y1;
        }
    }

    // ---- final hidden state: h_last[b, m] from last step's t0/t1 ----
    float* hout = out + (size_t)B_TOT * T_LEN + (size_t)b * H_DIM;
    #pragma unroll
    for (int k = 0; k < 8; ++k) {
        hout[k + 16 * half]     = t0[k];
        hout[k + 8 + 16 * half] = t1[k];
    }
}

extern "C" void kernel_launch(void* const* d_in, const int* in_sizes, int n_in,
                              void* d_out, int out_size, void* d_ws, size_t ws_size,
                              hipStream_t stream) {
    const float* x      = (const float*)d_in[0];
    const float* prev_h = (const float*)d_in[1];
    const float* W_ih   = (const float*)d_in[2];
    const float* W_hh   = (const float*)d_in[3];
    const float* b_ih   = (const float*)d_in[4];
    const float* b_hh   = (const float*)d_in[5];
    const float* Wd     = (const float*)d_in[6];
    const float* bd     = (const float*)d_in[7];
    float* out          = (float*)d_out;

    // 256 batch rows / 16 per wave = 16 independent waves, one per block
    rnn_wmma_kernel<<<dim3(B_TOT / 16), dim3(32), 0, stream>>>(
        x, prev_h, W_ih, W_hh, b_ih, b_hh, Wd, bd, out);
}
